// FSMNSeleNetV3_55327768707882
// MI455X (gfx1250) — compile-verified, matrix operand order checked by reading, not run on hardware
//
#include <hip/hip_runtime.h>

// ---------------------------------------------------------------------------
// DFSMN forward for MI455X (gfx1250, wave32, WMMA).
// All dense matmuls run through one tiled WMMA GEMM (f16 inputs, f32 accum):
//   C[M,N] = opt_relu(A[M,K] * W[N,K]^T + bias)
// M % 128 == 0 and K % 32 == 0 for every GEMM (layer-0 K=120 zero-padded to
// 128); only the decoder (N=9) takes an N-guarded instantiation.
// Block tile 128x128, 8 waves, wave tile 32x64 (2x4 WMMA 16x16x32 accums),
// double-buffered LDS with a software pipeline; packs via v_cvt_pk_rtz.
// FSMN memory block is a streaming register-ring-buffer kernel (reads s once).
// ---------------------------------------------------------------------------

typedef __attribute__((ext_vector_type(16))) _Float16 v16h;
typedef __attribute__((ext_vector_type(8)))  float    v8f;

#define LDS_K 36          // padded K stride (halfs): 8B aligned, bank-conflict relief
#define TILE_M 128
#define TILE_N 128
#define TILE_K 32

union Frag16 {
    v16h v;
    unsigned u[8];
};

__device__ inline unsigned pack2h(float a, float b) {
    auto p = __builtin_amdgcn_cvt_pkrtz(a, b);   // single v_cvt_pk_rtz_f16_f32
    return __builtin_bit_cast(unsigned, p);
}

// ---------------------------------------------------------------------------
// WMMA GEMM: 256 threads = 8 waves; block tile 128x128; wave tile 32x64.
// Preconditions: M % 128 == 0, K % 32 == 0, lda/ldb multiples of 4.
// ---------------------------------------------------------------------------
template<bool GUARD_N, bool HAS_BIAS, bool RELU>
__global__ __launch_bounds__(256)
void gemm_wmma_f16(const float* __restrict__ A, int lda,
                   const float* __restrict__ B, int ldb,
                   const float* __restrict__ bias,
                   float* __restrict__ C, int ldc,
                   int M, int N, int K)
{
    __shared__ _Float16 As[2][TILE_M * LDS_K];
    __shared__ _Float16 Bs[2][TILE_N * LDS_K];

    const int tid    = threadIdx.x;
    const int lane   = tid & 31;
    const int wid    = tid >> 5;
    const int wave_m = wid & 3;   // 0..3 -> 4 * 32 = 128 rows
    const int wave_n = wid >> 2;  // 0..1 -> 2 * 64 = 128 cols
    const int half   = lane >> 4; // 0/1
    const int lm     = lane & 15;

    const int blockRow = blockIdx.x * TILE_M;
    const int blockCol = blockIdx.y * TILE_N;

    // loader coordinates (constant per thread)
    const int lr  = tid >> 3;            // 0..31  base row
    const int lkc = (tid & 7) * 4;       // 0,4,...,28

    // fragment K map (ISA 7.12.2: 16-bit A 16x32; lanes 16..31 shifted by 8)
    int kks[8];
    #pragma unroll
    for (int v = 0; v < 8; ++v)
        kks[v] = ((v < 4) ? (2 * v) : (16 + 2 * (v - 4))) + half * 8;

    const int ra0 = (wave_m * 32 +  0 + lm) * LDS_K;
    const int ra1 = (wave_m * 32 + 16 + lm) * LDS_K;
    int cb[4];
    #pragma unroll
    for (int j = 0; j < 4; ++j)
        cb[j] = (wave_n * 64 + j * 16 + lm) * LDS_K;

    v8f acc[2][4];
    const v8f vzero = {0.f, 0.f, 0.f, 0.f, 0.f, 0.f, 0.f, 0.f};
    #pragma unroll
    for (int i = 0; i < 2; ++i)
        #pragma unroll
        for (int j = 0; j < 4; ++j)
            acc[i][j] = vzero;

    const int ktiles = K / TILE_K;

    float4 aR[4];          // staged A: rows lr + {0,32,64,96}
    float4 bR[4];          // staged B: rows lr + {0,32,64,96}

    auto load_global = [&](int kt) {
        const int k0 = kt * TILE_K;
        #pragma unroll
        for (int it = 0; it < 4; ++it) {
            const float* ap = A + (size_t)(blockRow + lr + it * 32) * lda + k0 + lkc;
            aR[it] = *(const float4*)ap;
            __builtin_prefetch(ap + TILE_K);     // global_prefetch_b8: tile kt+1
        }
        #pragma unroll
        for (int it = 0; it < 4; ++it) {
            const int gn = blockCol + lr + it * 32;
            if (!GUARD_N || gn < N)
                bR[it] = *(const float4*)(B + (size_t)gn * ldb + k0 + lkc);
            else
                bR[it] = float4{0.f, 0.f, 0.f, 0.f};
        }
    };

    auto store_lds = [&](int buf) {
        #pragma unroll
        for (int it = 0; it < 4; ++it) {
            _Float16* dst = &As[buf][(lr + it * 32) * LDS_K + lkc];
            *(unsigned*)(dst + 0) = pack2h(aR[it].x, aR[it].y);
            *(unsigned*)(dst + 2) = pack2h(aR[it].z, aR[it].w);
        }
        #pragma unroll
        for (int it = 0; it < 4; ++it) {
            _Float16* dst = &Bs[buf][(lr + it * 32) * LDS_K + lkc];
            *(unsigned*)(dst + 0) = pack2h(bR[it].x, bR[it].y);
            *(unsigned*)(dst + 2) = pack2h(bR[it].z, bR[it].w);
        }
    };

    load_global(0);
    store_lds(0);

    for (int kt = 0; kt < ktiles; ++kt) {
        __syncthreads();                 // stores to buf(kt&1) visible
        const int cur = kt & 1;

        if (kt + 1 < ktiles)
            load_global(kt + 1);         // overlap next tile's global loads

        Frag16 a0, a1, b[4];
        const _Float16* Ab = &As[cur][0];
        const _Float16* Bb = &Bs[cur][0];
        #pragma unroll
        for (int v = 0; v < 8; ++v) {
            const int kk = kks[v];
            a0.u[v] = *(const unsigned*)&Ab[ra0 + kk];
            a1.u[v] = *(const unsigned*)&Ab[ra1 + kk];
            #pragma unroll
            for (int j = 0; j < 4; ++j)
                b[j].u[v] = *(const unsigned*)&Bb[cb[j] + kk];
        }

        #pragma unroll
        for (int j = 0; j < 4; ++j) {
            acc[0][j] = __builtin_amdgcn_wmma_f32_16x16x32_f16(false, a0.v, false, b[j].v,
                                                               (short)0, acc[0][j], false, false);
            acc[1][j] = __builtin_amdgcn_wmma_f32_16x16x32_f16(false, a1.v, false, b[j].v,
                                                               (short)0, acc[1][j], false, false);
        }

        if (kt + 1 < ktiles)
            store_lds(cur ^ 1);          // fill the other buffer
    }

    // ---- epilogue: C layout = lanes 0..15 rows r, lanes 16..31 rows r+8
    #pragma unroll
    for (int wm = 0; wm < 2; ++wm) {
        #pragma unroll
        for (int wn = 0; wn < 4; ++wn) {
            const int col = blockCol + wave_n * 64 + wn * 16 + lm;
            float bv = 0.f;
            if (HAS_BIAS && (!GUARD_N || col < N)) bv = bias[col];
            #pragma unroll
            for (int r = 0; r < 8; ++r) {
                const int row = blockRow + wave_m * 32 + wm * 16 + half * 8 + r;
                float v = acc[wm][wn][r] + bv;
                if (RELU) v = fmaxf(v, 0.f);
                if (!GUARD_N || col < N)
                    C[(size_t)row * ldc + col] = v;
            }
        }
    }
}

// ---------------------------------------------------------------------------
// Fold channels into batch with K padding:
//   out[(b*4+n), t, 0:FP] = { x[b,t,n,0:F], zeros }      (FP = 128, F = 120)
// grid = (T, B*4), block = FP threads.  (assumes 4 channels)
// ---------------------------------------------------------------------------
__global__ __launch_bounds__(128)
void fold_channels_pad(const float* __restrict__ x, float* __restrict__ out,
                       int T, int F, int FP)
{
    const int t  = blockIdx.x;
    const int bn = blockIdx.y;
    const int b  = bn >> 2;
    const int n  = bn & 3;
    const int f  = threadIdx.x;
    float v = 0.f;
    if (f < F) v = x[(((size_t)b * T + t) * 4 + n) * F + f];
    out[((size_t)bn * T + t) * FP + f] = v;
}

// ---------------------------------------------------------------------------
// Zero-pad weight rows: dst[r, 0:KP] = { src[r, 0:K], zeros }
// grid = rows, block = KP threads.
// ---------------------------------------------------------------------------
__global__ __launch_bounds__(128)
void pad_rows(const float* __restrict__ src, float* __restrict__ dst, int K, int KP)
{
    const int r = blockIdx.x;
    const int k = threadIdx.x;
    dst[(size_t)r * KP + k] = (k < K) ? src[(size_t)r * K + k] : 0.f;
}

// ---------------------------------------------------------------------------
// FSMN memory block + residual, streaming over time with a register ring:
//   m[t] = s[t] + sum_{j=0..19} wl[d,j]*s[t-19+j] + wr[d,0]*s[t+1]
//   h[t] = (residual ? h[t] : 0) + m[t]     (in-place update of h)
// One thread per (bn, d); time split into chunks (grid.y). Coalesced along d.
// ---------------------------------------------------------------------------
__global__ __launch_bounds__(256)
void fsmn_mem_kernel(const float* __restrict__ s, float* __restrict__ h,
                     const float* __restrict__ wl, const float* __restrict__ wr,
                     int T, int D, int chunkT, int residual)
{
    const int bn = blockIdx.x;
    const int t0 = blockIdx.y * chunkT;
    const int t1 = (t0 + chunkT < T) ? (t0 + chunkT) : T;
    const int d  = threadIdx.x;            // blockDim.x == D == 256

    const float* sb = s + (size_t)bn * T * D + d;
    float*       hb = h + (size_t)bn * T * D + d;

    float tapL[20];
    #pragma unroll
    for (int j = 0; j < 20; ++j) tapL[j] = wl[d * 20 + j];
    const float tapR = wr[d];

    float past[19];                        // past[j] = s[t-19+j]
    #pragma unroll
    for (int j = 0; j < 19; ++j) {
        const int t = t0 - 19 + j;
        past[j] = (t >= 0) ? sb[(size_t)t * D] : 0.f;
    }

    float cur = (t0 < T) ? sb[(size_t)t0 * D] : 0.f;

    for (int t = t0; t < t1; ++t) {
        const float nxt = (t + 1 < T) ? sb[(size_t)(t + 1) * D] : 0.f;
        float accv = cur + tapL[19] * cur + tapR * nxt;
        #pragma unroll
        for (int j = 0; j < 19; ++j) accv += tapL[j] * past[j];
        const float res = residual ? (hb[(size_t)t * D] + accv) : accv;
        hb[(size_t)t * D] = res;
        #pragma unroll
        for (int j = 0; j < 18; ++j) past[j] = past[j + 1];
        past[18] = cur;
        cur = nxt;
    }
}

// ---------------------------------------------------------------------------
// Max-pool over the 4 channels: y[b,t,d] = max_n x2[(b*4+n), t, d]
// grid = (T, B), block = 256, D iterated by stride.
// ---------------------------------------------------------------------------
__global__ __launch_bounds__(256)
void maxpool_chan(const float* __restrict__ x2, float* __restrict__ y,
                  int T, int D)
{
    const int t = blockIdx.x;
    const int b = blockIdx.y;
    const float* base = x2 + (((size_t)b * 4) * T + t) * D;
    float* yo = y + ((size_t)b * T + t) * D;
    const size_t cs = (size_t)T * D;
    for (int d = threadIdx.x; d < D; d += blockDim.x) {
        float m = base[d];
        m = fmaxf(m, base[cs + d]);
        m = fmaxf(m, base[2 * cs + d]);
        m = fmaxf(m, base[3 * cs + d]);
        yo[d] = m;
    }
}

// ---------------------------------------------------------------------------
// Orchestration
// ---------------------------------------------------------------------------
extern "C" void kernel_launch(void* const* d_in, const int* in_sizes, int n_in,
                              void* d_out, int out_size, void* d_ws, size_t ws_size,
                              hipStream_t stream)
{
    (void)in_sizes; (void)n_in; (void)out_size;

    const float* x    = (const float*)d_in[0];   // [32,2000,4,120]
    const float* e0w  = (const float*)d_in[1];   // [512,120]
    const float* e0b  = (const float*)d_in[2];   // [512]
    const float* ew   = (const float*)d_in[3];   // [5,512,256]
    const float* eb   = (const float*)d_in[4];   // [5,512]
    const float* sw   = (const float*)d_in[5];   // [6,256,512]
    const float* cl   = (const float*)d_in[6];   // [6,256,20]
    const float* cr   = (const float*)d_in[7];   // [6,256,1]
    const float* e2w  = (const float*)d_in[8];   // [512,256]
    const float* e2b  = (const float*)d_in[9];   // [512]
    const float* dw   = (const float*)d_in[10];  // [9,512]
    const float* db   = (const float*)d_in[11];  // [9]
    float* out = (float*)d_out;                  // [32,2000,9]

    const int B = 32, T = 2000, F = 120, FP = 128;
    const int D1 = 512, P = 256, OUTD = 9;
    const int BN = B * 4;            // 128
    const int M  = BN * T;           // 256000 (multiple of 128)
    const int chunkT = 500;          // 4 time chunks per (bn) for the FSMN kernel

    // workspace layout (floats)
    const size_t nh = (size_t)BN * T * P;    //  65,536,000  h buffer
    const size_t ne = (size_t)BN * T * D1;   // 131,072,000  e / x2 buffer
    const size_t ns = (size_t)BN * T * P;    //  65,536,000  s / h0(padded) / y buffer
    const size_t nw = (size_t)D1 * FP;       //      65,536  padded expand0_w
    if (ws_size < (nh + ne + ns + nw) * sizeof(float)) return;

    float* hbuf = (float*)d_ws;
    float* ebuf = hbuf + nh;
    float* sbuf = ebuf + ne;
    float* wpad = sbuf + ns;

    const dim3 blk(256);
    auto gemm_grid = [](int m, int n) {
        return dim3((unsigned)(m / TILE_M),
                    (unsigned)((n + TILE_N - 1) / TILE_N));
    };

    // ---- layer 0 ----------------------------------------------------------
    fold_channels_pad<<<dim3((unsigned)T, (unsigned)BN), dim3(FP), 0, stream>>>(
        x, sbuf, T, F, FP);                                       // sbuf = h0 padded [128*2000,128]
    pad_rows<<<dim3((unsigned)D1), dim3(FP), 0, stream>>>(e0w, wpad, F, FP);

    gemm_wmma_f16<false, true, true><<<gemm_grid(M, D1), blk, 0, stream>>>(
        sbuf, FP, wpad, FP, e0b, ebuf, D1, M, D1, FP);            // e = relu(h0*W0^T + b0)

    gemm_wmma_f16<false, false, false><<<gemm_grid(M, P), blk, 0, stream>>>(
        ebuf, D1, sw + 0, D1, nullptr, sbuf, P, M, P, D1);        // s = e * Ws0^T

    fsmn_mem_kernel<<<dim3((unsigned)BN, (unsigned)(T / chunkT)), blk, 0, stream>>>(
        sbuf, hbuf, cl + 0, cr + 0, T, P, chunkT, /*residual=*/0);

    // ---- layers 1..5 ------------------------------------------------------
    for (int i = 1; i < 6; ++i) {
        gemm_wmma_f16<false, true, true><<<gemm_grid(M, D1), blk, 0, stream>>>(
            hbuf, P, ew + (size_t)(i - 1) * D1 * P, P,
            eb + (size_t)(i - 1) * D1, ebuf, D1, M, D1, P);       // e = relu(h*We^T + be)

        gemm_wmma_f16<false, false, false><<<gemm_grid(M, P), blk, 0, stream>>>(
            ebuf, D1, sw + (size_t)i * P * D1, D1, nullptr, sbuf, P, M, P, D1);

        fsmn_mem_kernel<<<dim3((unsigned)BN, (unsigned)(T / chunkT)), blk, 0, stream>>>(
            sbuf, hbuf, cl + (size_t)i * P * 20, cr + (size_t)i * P,
            T, P, chunkT, /*residual=*/1);
    }

    // ---- head -------------------------------------------------------------
    gemm_wmma_f16<false, true, true><<<gemm_grid(M, D1), blk, 0, stream>>>(
        hbuf, P, e2w, P, e2b, ebuf, D1, M, D1, P);                // x2 = relu(h*W2^T + b2)

    maxpool_chan<<<dim3((unsigned)T, (unsigned)B), blk, 0, stream>>>(
        ebuf, sbuf, T, D1);                                       // y = max over channels

    gemm_wmma_f16<true, true, false><<<gemm_grid(B * T, OUTD), blk, 0, stream>>>(
        sbuf, D1, dw, D1, db, out, OUTD, B * T, OUTD, D1);        // z = y*dec^T + db
}
